// SparseAttention_42185168782027
// MI455X (gfx1250) — matrix operand order repaired and checked by reading, not compiled, not run on hardware
//
#include <hip/hip_runtime.h>

#define BB 8
#define NN 2048
#define CC 1024
#define KKEEP 1638
#define SCALE 0.03125f  // 1/sqrt(1024)

typedef __bf16 bf16_t;
typedef __bf16 v8bf  __attribute__((ext_vector_type(8)));
typedef __bf16 v16bf __attribute__((ext_vector_type(16)));
typedef float  v8f   __attribute__((ext_vector_type(8)));

// Build a 16-element bf16 fragment from two 16-byte (8 x bf16) loads.
static __device__ __forceinline__ v16bf frag16(const bf16_t* lo, const bf16_t* hi) {
    v8bf a = *(const v8bf*)lo;
    v8bf b = *(const v8bf*)hi;
    return __builtin_shufflevector(a, b, 0,1,2,3,4,5,6,7,8,9,10,11,12,13,14,15);
}

static __device__ __forceinline__ v8f wmma_bf16(v16bf a, v16bf b, v8f c) {
    // D = A(16x32 bf16) * B(32x16 bf16) + C(16x16 f32)
    return __builtin_amdgcn_wmma_f32_16x16x32_bf16(false, a, false, b, (short)0, c,
                                                   false, false);
}

// 8-tile (16 x 128 output) WMMA GEMM slice with explicit double buffering.
// Two live fragment arrays (bb0/bb1) force the allocator to keep the next
// k-step's 18 loads in flight while the current 8 WMMAs execute.
// Arow: per-lane A row base (row = lane%16). Bbase: per-lane B col base
// (col = lane%16 of tile 0); tiles are btile_stride elements apart.
static __device__ __forceinline__ void gemm_8tiles(
    const bf16_t* __restrict__ Arow, const bf16_t* __restrict__ Bbase,
    size_t btile_stride, int klen, int asel, int bsel, v8f acc[8]) {
    v16bf bb0[8], bb1[8];
    #pragma unroll
    for (int t = 0; t < 8; ++t) {
        const bf16_t* p = Bbase + (size_t)t * btile_stride + bsel;
        bb0[t] = frag16(p, p + 16);
    }
    v16bf a_cur = frag16(Arow + asel, Arow + asel + 16);
    for (int kk = 0; kk < klen; kk += 64) {
        const int k1 = kk + 32;
        #pragma unroll
        for (int t = 0; t < 8; ++t) {
            const bf16_t* p = Bbase + (size_t)t * btile_stride + k1 + bsel;
            bb1[t] = frag16(p, p + 16);
        }
        v16bf a_n1 = frag16(Arow + k1 + asel, Arow + k1 + asel + 16);
        #pragma unroll
        for (int t = 0; t < 8; ++t) acc[t] = wmma_bf16(a_cur, bb0[t], acc[t]);

        const int k2 = (kk + 64 < klen) ? kk + 64 : 0;  // last iter: harmless reload
        #pragma unroll
        for (int t = 0; t < 8; ++t) {
            const bf16_t* p = Bbase + (size_t)t * btile_stride + k2 + bsel;
            bb0[t] = frag16(p, p + 16);
        }
        v16bf a_n2 = frag16(Arow + k2 + asel, Arow + k2 + asel + 16);
        #pragma unroll
        for (int t = 0; t < 8; ++t) acc[t] = wmma_bf16(a_n1, bb1[t], acc[t]);
        a_cur = a_n2;
    }
}

// ------------------------------------------------------------- f32 -> bf16
__global__ void conv_kernel(const float* __restrict__ src, bf16_t* __restrict__ dst,
                            int n) {
    int i = blockIdx.x * blockDim.x + threadIdx.x;
    if (i < n) dst[i] = (bf16_t)src[i];
}

// ---------------------------------------------------------------- QKV GEMM
// Y[m,d] = sum_c x[m,c] * W[d,c] + bias[d]   (torch Linear)
// W in original [d][c] row-major bf16: B-fragments are contiguous. All
// fragments straight from global (L2-resident); double-buffered pipeline.
// blockIdx.y: 0 -> Q (row-major), 1 -> K (row-major), 2 -> V (transposed [b][d][n])
__global__ void __launch_bounds__(256)
qkv_gemm_kernel(const bf16_t* __restrict__ xb,
                const bf16_t* __restrict__ wq, const bf16_t* __restrict__ wk,
                const bf16_t* __restrict__ wv,
                const float* __restrict__ bq, const float* __restrict__ bk,
                const float* __restrict__ bv,
                bf16_t* __restrict__ Qb, bf16_t* __restrict__ Kb,
                bf16_t* __restrict__ Vt) {
    const int which = blockIdx.y;
    const bf16_t* W    = (which == 0) ? wq : (which == 1) ? wk : wv;
    const float*  bias = (which == 0) ? bq : (which == 1) ? bk : bv;

    const int m0   = blockIdx.x * 16;       // 16-row slab; 8 waves cover 1024 cols
    const int tid  = threadIdx.x;
    const int lane = tid & 31;
    const int wave = tid >> 5;
    const int lrow = lane & 15;
    const int asel  = (lane < 16) ? 0 : 8;
    const int bsel  = (lane < 16) ? 0 : 8;
    const int rbase = (lane < 16) ? 0 : 8;

    const bf16_t* Arow  = xb + (size_t)(m0 + lrow) * CC;
    const bf16_t* Wwave = W + (size_t)(wave * 128 + lrow) * CC;

    v8f acc[8];
    #pragma unroll
    for (int t = 0; t < 8; ++t) acc[t] = (v8f){};

    gemm_8tiles(Arow, Wwave, (size_t)16 * CC, CC, asel, bsel, acc);

    #pragma unroll
    for (int t = 0; t < 8; ++t) {
        int d = wave * 128 + t * 16 + lrow;
        float bv_ = bias[d];
        #pragma unroll
        for (int r = 0; r < 8; ++r) {
            float v = acc[t][r] + bv_;
            int mrow = m0 + rbase + r;
            if (which == 0) {
                Qb[(size_t)mrow * CC + d] = (bf16_t)v;
            } else if (which == 1) {
                Kb[(size_t)mrow * CC + d] = (bf16_t)v;
            } else {
                int bidx = mrow >> 11;        // / NN
                int nidx = mrow & (NN - 1);
                Vt[((size_t)bidx * CC + d) * NN + nidx] = (bf16_t)v;
            }
        }
    }
}

// ---------------------------------------------------------- fused attention
// One block = 16 query rows of one batch. 8 waves.
// Phase 1: S = (Q K^T)*scale -> LDS (f32, 16x2048), two 8-tile column groups
// Phase 2: per-row top-k threshold via bisection, masked softmax -> P (bf16)
// Phase 3: out = P @ V (V pre-transposed [b][d][m])
__global__ void __launch_bounds__(256)
attn_kernel(const bf16_t* __restrict__ Qb, const bf16_t* __restrict__ Kb,
            const bf16_t* __restrict__ Vt, float* __restrict__ out) {
    extern __shared__ char smem[];
    const int SSTR = NN + 4;   // f32 row stride (pad)
    const int PSTR = NN + 8;   // bf16 row stride (pad, keeps 16B alignment)
    float*  S = (float*)smem;
    bf16_t* P = (bf16_t*)(smem + (size_t)16 * SSTR * sizeof(float));

    const int b    = blockIdx.y;
    const int n0   = blockIdx.x * 16;
    const int tid  = threadIdx.x;
    const int lane = tid & 31;
    const int wave = tid >> 5;
    const int lrow = lane & 15;
    const int asel  = (lane < 16) ? 0 : 8;
    const int bsel  = (lane < 16) ? 0 : 8;
    const int rbase = (lane < 16) ? 0 : 8;

    const bf16_t* Qrow  = Qb + ((size_t)b * NN + n0 + lrow) * CC;
    const bf16_t* Kwave = Kb + (size_t)b * NN * CC + (size_t)(wave * 256 + lrow) * CC;

    // ---- Phase 1: scores. Wave owns 256 key columns = 2 groups x 8 tiles.
    for (int g = 0; g < 2; ++g) {
        v8f acc[8];
        #pragma unroll
        for (int t = 0; t < 8; ++t) acc[t] = (v8f){};

        gemm_8tiles(Qrow, Kwave + (size_t)g * 128 * CC, (size_t)16 * CC, CC,
                    asel, bsel, acc);

        #pragma unroll
        for (int t = 0; t < 8; ++t) {
            int col = wave * 256 + g * 128 + t * 16 + lrow;
            #pragma unroll
            for (int r = 0; r < 8; ++r)
                S[(rbase + r) * SSTR + col] = acc[t][r] * SCALE;
        }
    }
    __syncthreads();

    // ---- Phase 2: threshold (keep top KKEEP) + softmax. Wave w -> rows 2w,2w+1.
    for (int rr = 0; rr < 2; ++rr) {
        int row = wave * 2 + rr;
        float* Srow = S + row * SSTR;
        float vmax = -3.4e38f, vmin = 3.4e38f;
        for (int i = lane; i < NN; i += 32) {
            float s = Srow[i];
            vmax = fmaxf(vmax, s);
            vmin = fminf(vmin, s);
        }
        #pragma unroll
        for (int off = 16; off > 0; off >>= 1) {
            vmax = fmaxf(vmax, __shfl_xor(vmax, off, 32));
            vmin = fminf(vmin, __shfl_xor(vmin, off, 32));
        }
        // bisection: find t with count(s >= t) ~= KKEEP
        float lo = vmin, hi = vmax;
        for (int it = 0; it < 22; ++it) {
            float mid = 0.5f * (lo + hi);
            int cnt = 0;
            for (int i = lane; i < NN; i += 32) cnt += (Srow[i] >= mid) ? 1 : 0;
            #pragma unroll
            for (int off = 16; off > 0; off >>= 1) cnt += __shfl_xor(cnt, off, 32);
            if (cnt > KKEEP) lo = mid; else hi = mid;
        }
        float thr = hi;  // count(>= hi) <= KKEEP, includes the row max
        float sum = 0.f;
        for (int i = lane; i < NN; i += 32) {
            float s = Srow[i];
            float p = (s >= thr) ? __expf(s - vmax) : 0.0f;  // masked -> exp(-1e10)~0
            Srow[i] = p;
            sum += p;
        }
        #pragma unroll
        for (int off = 16; off > 0; off >>= 1) sum += __shfl_xor(sum, off, 32);
        float inv = 1.0f / sum;
        bf16_t* Prow = P + row * PSTR;
        for (int i = lane; i < NN; i += 32) Prow[i] = (bf16_t)(Srow[i] * inv);
    }
    __syncthreads();

    // ---- Phase 3: out = P @ V.  Each wave owns 128 feature cols = 8 tiles.
    {
        const bf16_t* Vwave = Vt + (size_t)b * CC * NN + (size_t)(wave * 128 + lrow) * NN;
        const bf16_t* Pa = P + lrow * PSTR;   // LDS (A side)
        v8f oacc[8];
        #pragma unroll
        for (int t = 0; t < 8; ++t) oacc[t] = (v8f){};

        gemm_8tiles(Pa, Vwave, (size_t)16 * NN, NN, asel, bsel, oacc);

        #pragma unroll
        for (int t = 0; t < 8; ++t) {
            int d = wave * 128 + t * 16 + lrow;
            #pragma unroll
            for (int r = 0; r < 8; ++r)
                out[((size_t)b * NN + n0 + rbase + r) * CC + d] = oacc[t][r];
        }
    }
}

// ---------------------------------------------------------------- launch
extern "C" void kernel_launch(void* const* d_in, const int* in_sizes, int n_in,
                              void* d_out, int out_size, void* d_ws, size_t ws_size,
                              hipStream_t stream) {
    const float* x  = (const float*)d_in[0];
    const float* Wq = (const float*)d_in[1];
    const float* bq = (const float*)d_in[2];
    const float* Wk = (const float*)d_in[3];
    const float* bk = (const float*)d_in[4];
    const float* Wv = (const float*)d_in[5];
    const float* bv = (const float*)d_in[6];

    const size_t nx = (size_t)BB * NN * CC;  // 16,777,216
    const size_t nw = (size_t)CC * CC;       // 1,048,576
    bf16_t* xb  = (bf16_t*)d_ws;
    bf16_t* wqb = xb + nx;
    bf16_t* wkb = wqb + nw;
    bf16_t* wvb = wkb + nw;
    bf16_t* Qb  = wvb + nw;
    bf16_t* Kb  = Qb + nx;
    bf16_t* Vt  = Kb + nx;

    conv_kernel<<<(int)(nx / 256), 256, 0, stream>>>(x, xb, (int)nx);
    conv_kernel<<<(int)(nw / 256), 256, 0, stream>>>(Wq, wqb, (int)nw);
    conv_kernel<<<(int)(nw / 256), 256, 0, stream>>>(Wk, wkb, (int)nw);
    conv_kernel<<<(int)(nw / 256), 256, 0, stream>>>(Wv, wvb, (int)nw);

    // 16-row slabs x {Q,K,V}: one fused launch, fragments straight from L2.
    qkv_gemm_kernel<<<dim3(BB * NN / 16, 3), 256, 0, stream>>>(
        xb, wqb, wkb, wvb, bq, bk, bv, Qb, Kb, Vt);

    size_t smem = (size_t)16 * (NN + 4) * sizeof(float)
                + (size_t)16 * (NN + 8) * sizeof(bf16_t);   // ~197 KB of 320 KB/WGP
    hipFuncSetAttribute((const void*)attn_kernel,
                        hipFuncAttributeMaxDynamicSharedMemorySize, (int)smem);
    attn_kernel<<<dim3(NN / 16, BB), 256, smem, stream>>>(Qb, Kb, Vt, (float*)d_out);
}
